// EfficientLoFTRAggregatedAttention_73607149519169
// MI455X (gfx1250) — compile-verified, matrix operand order checked by reading, not run on hardware
//
#include <hip/hip_runtime.h>
#include <hip/hip_bf16.h>

typedef _Float16 h16;
typedef __attribute__((ext_vector_type(16))) _Float16 v16h;
typedef __attribute__((ext_vector_type(8)))  _Float16 v8h;
typedef __attribute__((ext_vector_type(8)))  float    v8f;
typedef __attribute__((ext_vector_type(4)))  unsigned int u32x4;
typedef __attribute__((ext_vector_type(8)))  unsigned int u32x8;

#define NB 4
#define NC 256
#define NH 160
#define NW 160
#define AH 40
#define AW 40
#define NS 1600          // AH*AW
#define NTOK (NB*NS)     // 6400
#define NPIX (NB*NH*NW)  // 102400
#define NHEAD 8
#define HD 32
#define LN_EPS 1e-5f

// ---------------------------------------------------------------------------
// WMMA helpers. CDNA5 16-bit A (16xK=32) per-lane element mapping:
//   lane l: row m = l&15, half hl = l>>4
//   elements 0..7  -> K = 8*hl   .. 8*hl+7    (contiguous)
//   elements 8..15 -> K = 16+8*hl .. 23+8*hl  (contiguous)
// B (32x16) mirrors this with lane -> column n. Both become two b128 loads
// from a row-major row when the "row" axis is (m for A / n for Bt).
// ---------------------------------------------------------------------------
__device__ __forceinline__ v16h load_frag(const h16* __restrict__ row, int hl) {
  v8h lo = *(const v8h*)(row + 8 * hl);
  v8h hi = *(const v8h*)(row + 16 + 8 * hl);
  v16h f;
#pragma unroll
  for (int i = 0; i < 8; ++i) { f[i] = lo[i]; f[i + 8] = hi[i]; }
  return f;
}

__device__ __forceinline__ v8f wmma_f16(v16h a, v16h b, v8f c) {
  return __builtin_amdgcn_wmma_f32_16x16x32_f16(false, a, false, b, (short)0, c,
                                                false, false);
}

// ---------------------------------------------------------------------------
// Tensor Data Mover: DMA a 2D f16 tile (tile_n rows x tile_k elements,
// row stride stride_k elements) from global memory into LDS at lds_addr.
// D# per CDNA5 ISA ch.8: group0 = {count/type/addresses}, group1 = dims.
// 2-group form (VADDR2/3 omitted) is the <=2D tensor encoding.
// Tracked with TENSORcnt; wave-ordered, so s_wait_tensorcnt N guarantees the
// oldest N outstanding transfers... (wait <=1 -> oldest of 2 complete).
// ---------------------------------------------------------------------------
__device__ __forceinline__ void tdm_load_tile_f16(unsigned lds_addr,
                                                  const void* gaddr,
                                                  unsigned tile_k,
                                                  unsigned tile_n,
                                                  unsigned stride_k) {
  unsigned long long ga = (unsigned long long)(size_t)gaddr;
  u32x4 g0;
  g0.x = 1u;                                    // count=1 (valid descriptor)
  g0.y = lds_addr;                              // LDS byte address
  g0.z = (unsigned)ga;                          // global_addr[31:0]
  g0.w = (unsigned)(ga >> 32) | 0x80000000u;    // global_addr[56:32] | type=2
  u32x8 g1;
  g1[0] = 0x00010000u;                          // data_size=1 (2B); mask/pad=0
  g1[1] = (tile_k & 0xffffu) << 16;             // tensor_dim0[15:0]
  g1[2] = (tile_k >> 16) | ((tile_n & 0xffffu) << 16);  // d0[31:16] | d1[15:0]
  g1[3] = (tile_n >> 16) | ((tile_k & 0xffffu) << 16);  // d1[31:16] | tile_dim0
  g1[4] = tile_n & 0xffffu;                     // tile_dim1 (tile_dim2=0)
  g1[5] = stride_k;                             // tensor_dim0_stride[31:0]
  g1[6] = 0u;                                   // stride hi / dim1_stride = 0
  g1[7] = 0u;
  asm volatile("tensor_load_to_lds %0, %1" :: "s"(g0), "s"(g1) : "memory");
}

// ---------------------------------------------------------------------------
// Weight prep: W [K][N] f32 -> Wt [N][K] f16
// ---------------------------------------------------------------------------
__global__ void k_wt(const float* __restrict__ in, h16* __restrict__ out,
                     int K, int N) {
  int n = blockIdx.x;
  for (int k = threadIdx.x; k < K; k += blockDim.x)
    out[(size_t)n * K + k] = (h16)in[(size_t)k * N + n];
}

// ---------------------------------------------------------------------------
// Aggregation: depthwise 4x4/s4 conv + 4x4 maxpool. One block = (b,c,window-row)
// coalesced row reads, 4-lane shuffle reduction per window.
// ---------------------------------------------------------------------------
__global__ __launch_bounds__(160)
void k_agg(const float* __restrict__ x, const float* __restrict__ w,
           float* __restrict__ qmap, float* __restrict__ kvmap) {
  int bc = blockIdx.x;            // b*C + c
  int wy = blockIdx.y;            // 0..39
  int c  = bc & 255;
  int xc = threadIdx.x;           // 0..159
  const float* base = x + ((size_t)bc * NH + wy * 4) * NW + xc;
  int j = xc & 3;
  float conv = 0.f, mx = -INFINITY;
#pragma unroll
  for (int i = 0; i < 4; ++i) {
    float v = base[i * NW];
    conv += v * w[c * 16 + i * 4 + j];
    mx = fmaxf(mx, v);
  }
#pragma unroll
  for (int off = 1; off < 4; off <<= 1) {
    conv += __shfl_xor(conv, off);
    mx = fmaxf(mx, __shfl_xor(mx, off));
  }
  if (j == 0) {
    int b = bc >> 8;
    size_t tok = (size_t)b * NS + wy * AW + (xc >> 2);
    qmap[tok * NC + c]  = conv;
    kvmap[tok * NC + c] = mx;
  }
}

// ---------------------------------------------------------------------------
// LayerNorm over C=256, emit f16. blockIdx.y selects q/kv map.
// ---------------------------------------------------------------------------
__global__ __launch_bounds__(256)
void k_lnh(const float* __restrict__ qmap, const float* __restrict__ kvmap,
           const float* __restrict__ gamma, const float* __restrict__ beta,
           h16* __restrict__ q_in, h16* __restrict__ kv_in) {
  const float* in = blockIdx.y ? kvmap : qmap;
  h16* out        = blockIdx.y ? kv_in : q_in;
  size_t tok = blockIdx.x;
  int t = threadIdx.x;
  float v = in[tok * NC + t];
  float s = v, s2 = v * v;
#pragma unroll
  for (int off = 1; off < 32; off <<= 1) {
    s += __shfl_xor(s, off);
    s2 += __shfl_xor(s2, off);
  }
  __shared__ float ws1[8], ws2[8];
  if ((t & 31) == 0) { ws1[t >> 5] = s; ws2[t >> 5] = s2; }
  __syncthreads();
  float ts = 0.f, ts2 = 0.f;
#pragma unroll
  for (int i = 0; i < 8; ++i) { ts += ws1[i]; ts2 += ws2[i]; }
  float mu  = ts * (1.f / NC);
  float var = ts2 * (1.f / NC) - mu * mu;
  float rstd = rsqrtf(var + LN_EPS);
  out[tok * NC + t] = (h16)((v - mu) * rstd * gamma[t] + beta[t]);
}

// ---------------------------------------------------------------------------
// Generic WMMA GEMM: C[M][N] f32 = A[M][K] f16 x Wt[N][K] f16.
// One wave per block; wave computes a 16x64 tile (4 WMMAs / K-step).
// B tiles (64 rows x 32 k) are DMA'd by the TDM into a double-buffered LDS
// slab one K-step ahead; the wave consumes the previous slab via ds_load_b128.
// ---------------------------------------------------------------------------
__global__ __launch_bounds__(32)
void k_gemm(const h16* __restrict__ A, const h16* __restrict__ Bt,
            float* __restrict__ C, int M, int N, int K) {
  __shared__ __align__(16) h16 slab[2][64 * 32];
  int m0 = blockIdx.x * 16;
  int n0 = blockIdx.y * 64;
  int lane = threadIdx.x;
  int l15 = lane & 15, hl = lane >> 4;
  const h16* arow  = A + (size_t)(m0 + l15) * K;
  const h16* bbase = Bt + (size_t)n0 * K;
  unsigned lds0 = (unsigned)(size_t)(const void*)&slab[0][0];
  unsigned lds1 = (unsigned)(size_t)(const void*)&slab[1][0];
  v8f acc[4] = {};
  tdm_load_tile_f16(lds0, bbase, 32u, 64u, (unsigned)K);
  int cur = 0;
  for (int k0 = 0; k0 < K; k0 += 32) {
    if (k0 + 32 < K) {
      tdm_load_tile_f16(cur ? lds0 : lds1, bbase + (k0 + 32), 32u, 64u,
                        (unsigned)K);
      __builtin_amdgcn_s_wait_tensorcnt(1);   // oldest (current slab) done
    } else {
      __builtin_amdgcn_s_wait_tensorcnt(0);
    }
    asm volatile("" ::: "memory");
    v16h af = load_frag(arow + k0, hl);
    const h16* sb = &slab[cur][0];
#pragma unroll
    for (int jj = 0; jj < 4; ++jj) {
      v16h bf = load_frag(sb + (jj * 16 + l15) * 32, hl);
      acc[jj] = wmma_f16(af, bf, acc[jj]);
    }
    cur ^= 1;
  }
  __shared__ __align__(16) float s[16 * 64];
#pragma unroll
  for (int jj = 0; jj < 4; ++jj)
#pragma unroll
    for (int r = 0; r < 8; ++r)
      s[(r + 8 * hl) * 64 + jj * 16 + l15] = acc[jj][r];
  __syncthreads();
#pragma unroll
  for (int r = 0; r < 16; ++r) {
    float2 v = ((const float2*)(s + r * 64))[lane];
    *(float2*)(C + (size_t)(m0 + r) * N + n0 + 2 * lane) = v;
  }
}

// ---------------------------------------------------------------------------
// RoPE + layout repack: qs/ks/vs f32 [tok][256] -> Qh/Kh f16 [B][8][S][32],
// Vt f16 [B][8][32][S]. rotate_half pairs (c, c+128).
// ---------------------------------------------------------------------------
__global__ __launch_bounds__(128)
void k_rope(const float* __restrict__ qs, const float* __restrict__ ks,
            const float* __restrict__ vs, const float* __restrict__ cosm,
            const float* __restrict__ sinm, h16* __restrict__ Qh,
            h16* __restrict__ Kh, h16* __restrict__ Vt) {
  size_t tok = blockIdx.x;
  int b = (int)(tok / NS), sp = (int)(tok % NS);
  int t = threadIdx.x;           // 0..127
  int c1 = t, c2 = t + 128;
  size_t cs1 = (size_t)(b * NC + c1) * NS + sp;
  size_t cs2 = (size_t)(b * NC + c2) * NS + sp;
  float co1 = cosm[cs1], si1 = sinm[cs1];
  float co2 = cosm[cs2], si2 = sinm[cs2];
  size_t tb = tok * NC;
  float q1 = qs[tb + c1], q2 = qs[tb + c2];
  float k1 = ks[tb + c1], k2 = ks[tb + c2];
  float qo1 = q1 * co1 - q2 * si1;
  float qo2 = q2 * co2 + q1 * si2;
  float ko1 = k1 * co1 - k2 * si1;
  float ko2 = k2 * co2 + k1 * si2;
  int h1 = c1 >> 5, d1 = c1 & 31;
  int h2 = c2 >> 5, d2 = c2 & 31;
  size_t a1 = ((size_t)(b * NHEAD + h1) * NS + sp) * HD + d1;
  size_t a2 = ((size_t)(b * NHEAD + h2) * NS + sp) * HD + d2;
  Qh[a1] = (h16)qo1; Qh[a2] = (h16)qo2;
  Kh[a1] = (h16)ko1; Kh[a2] = (h16)ko2;
  size_t v1a = ((size_t)(b * NHEAD + h1) * HD + d1) * NS + sp;
  size_t v2a = ((size_t)(b * NHEAD + h2) * HD + d2) * NS + sp;
  Vt[v1a] = (h16)vs[tb + c1];
  Vt[v2a] = (h16)vs[tb + c2];
}

// ---------------------------------------------------------------------------
// Flash attention. Block = 4 waves x 16 q-rows; loop K tokens in chunks of 32.
// head_dim=32 == WMMA K -> one WMMA per 16x16 score tile.
// ---------------------------------------------------------------------------
__global__ __launch_bounds__(128)
void k_attn(const h16* __restrict__ Qh, const h16* __restrict__ Kh,
            const h16* __restrict__ Vt, h16* __restrict__ ctx) {
  int bh = blockIdx.y;               // b*8+h
  int b = bh >> 3, hd = bh & 7;
  int w = threadIdx.x >> 5;
  int lane = threadIdx.x & 31;
  int l15 = lane & 15, hl = lane >> 4;
  int q0 = blockIdx.x * 64 + w * 16;
  const h16* Qb = Qh + (size_t)bh * NS * HD;
  const h16* Kb = Kh + (size_t)bh * NS * HD;
  const h16* Vb = Vt + (size_t)bh * HD * NS;
  __shared__ __align__(16) h16 ps[4 * 16 * 32];
  h16* pm = ps + w * 512;
  v16h qf = load_frag(Qb + (size_t)(q0 + l15) * HD, hl);
  v8f c0v = {}, c1v = {};
  float mrow[8], lrow[8];
#pragma unroll
  for (int r = 0; r < 8; ++r) { mrow[r] = -INFINITY; lrow[r] = 0.f; }
  const float scale = 0.17677669529663689f;  // 1/sqrt(32)
  for (int kk = 0; kk < NS; kk += 32) {
    v16h kf0 = load_frag(Kb + (size_t)(kk + l15) * HD, hl);
    v16h kf1 = load_frag(Kb + (size_t)(kk + 16 + l15) * HD, hl);
    v8f z = {};
    v8f s0 = wmma_f16(qf, kf0, z);
    v8f s1 = wmma_f16(qf, kf1, z);
#pragma unroll
    for (int r = 0; r < 8; ++r) {
      float a0 = s0[r] * scale, a1 = s1[r] * scale;
      float m = fmaxf(a0, a1);
#pragma unroll
      for (int off = 1; off < 16; off <<= 1) m = fmaxf(m, __shfl_xor(m, off));
      float nm = fmaxf(mrow[r], m);
      float corr = __expf(mrow[r] - nm);
      mrow[r] = nm;
      float p0 = __expf(a0 - nm), p1 = __expf(a1 - nm);
      float rs = p0 + p1;
#pragma unroll
      for (int off = 1; off < 16; off <<= 1) rs += __shfl_xor(rs, off);
      lrow[r] = lrow[r] * corr + rs;
      c0v[r] *= corr; c1v[r] *= corr;
      pm[(r + 8 * hl) * 32 + l15]      = (h16)p0;
      pm[(r + 8 * hl) * 32 + 16 + l15] = (h16)p1;
    }
    __syncthreads();
    v16h pf  = load_frag(pm + l15 * 32, hl);
    v16h vf0 = load_frag(Vb + (size_t)l15 * NS + kk, hl);
    v16h vf1 = load_frag(Vb + (size_t)(16 + l15) * NS + kk, hl);
    c0v = wmma_f16(pf, vf0, c0v);
    c1v = wmma_f16(pf, vf1, c1v);
    __syncthreads();
  }
#pragma unroll
  for (int r = 0; r < 8; ++r) {
    float inv = 1.f / lrow[r];
    int row = q0 + r + 8 * hl;
    size_t ob = ((size_t)b * NS + row) * NC + hd * HD;
    ctx[ob + l15]      = (h16)(c0v[r] * inv);
    ctx[ob + 16 + l15] = (h16)(c1v[r] * inv);
  }
}

// ---------------------------------------------------------------------------
// Fully fused MLP: per block = 16 pixels (same y, x0..x0+15).
// Stage concat(hidden, bilinear(message)) f16 in LDS -> GEMM1(512x512)+leaky
// -> LDS -> GEMM2(512x256) -> LayerNorm -> residual -> NCHW store.
// 8 waves; WMMA A-frags come from LDS, B-frags from global (+prefetch ahead).
// ---------------------------------------------------------------------------
__global__ __launch_bounds__(256)
void k_mlp(const float* __restrict__ xin, const float* __restrict__ msg,
           const h16* __restrict__ w1t, const h16* __restrict__ w2t,
           const float* __restrict__ gamma, const float* __restrict__ beta,
           float* __restrict__ out) {
  __shared__ __align__(16) h16 sA[16 * 512];   // concat input tile
  __shared__ __align__(16) h16 sB[16 * 512];   // activations
  __shared__ float sC[16 * 260];               // fc2 output (padded stride)
  __shared__ float smu[16], srstd[16];

  int m0 = blockIdx.x * 16;                    // 16 consecutive pixels
  int b  = m0 / (NH * NW);
  int p  = m0 % (NH * NW);
  int y  = p / NW, x0 = p % NW;
  int t  = threadIdx.x;                        // 0..255 (c index for staging)

  // --- stage hidden channels (c = t), 16 consecutive x per thread ---
  {
    const float* hp = xin + ((size_t)(b * NC + t) * NH + y) * NW + x0;
#pragma unroll
    for (int pp = 0; pp < 16; ++pp) sA[pp * 512 + t] = (h16)hp[pp];
  }
  // --- stage bilinear-upsampled message (half-pixel, scale 1/4) ---
  {
    const float* mb = msg + (size_t)b * NS * NC;
    float sy = y * 0.25f - 0.375f;
    float fy = floorf(sy);
    float wy = sy - fy;
    int y0 = (int)fy; int y1 = y0 + 1;
    y0 = y0 < 0 ? 0 : (y0 > AH - 1 ? AH - 1 : y0);
    y1 = y1 < 0 ? 0 : (y1 > AH - 1 ? AH - 1 : y1);
#pragma unroll
    for (int pp = 0; pp < 16; ++pp) {
      int xg = x0 + pp;
      float sx = xg * 0.25f - 0.375f;
      float fx = floorf(sx);
      float wx = sx - fx;
      int xa = (int)fx; int xb = xa + 1;
      xa = xa < 0 ? 0 : (xa > AW - 1 ? AW - 1 : xa);
      xb = xb < 0 ? 0 : (xb > AW - 1 ? AW - 1 : xb);
      float v0 = mb[(size_t)(y0 * AW + xa) * NC + t] * (1.f - wx) +
                 mb[(size_t)(y0 * AW + xb) * NC + t] * wx;
      float v1 = mb[(size_t)(y1 * AW + xa) * NC + t] * (1.f - wx) +
                 mb[(size_t)(y1 * AW + xb) * NC + t] * wx;
      sA[pp * 512 + 256 + t] = (h16)(v0 * (1.f - wy) + v1 * wy);
    }
  }
  __syncthreads();

  int w = t >> 5, lane = t & 31;
  int l15 = lane & 15, hl = lane >> 4;

  // --- GEMM1: [16,512] x [512,512] + leaky relu -> sB ---
  {
    v8f acc[4] = {};
    for (int k0 = 0; k0 < 512; k0 += 32) {
      v16h af = load_frag(sA + l15 * 512 + k0, hl);
#pragma unroll
      for (int jj = 0; jj < 4; ++jj) {
        const h16* brow = w1t + (size_t)(w * 64 + jj * 16 + l15) * 512 + k0;
        if (k0 + 32 < 512) __builtin_prefetch(brow + 32, 0, 1);
        v16h bf = load_frag(brow, hl);
        acc[jj] = wmma_f16(af, bf, acc[jj]);
      }
    }
#pragma unroll
    for (int jj = 0; jj < 4; ++jj)
#pragma unroll
      for (int r = 0; r < 8; ++r) {
        float xv = acc[jj][r];
        xv = xv >= 0.f ? xv : 0.01f * xv;
        sB[(r + 8 * hl) * 512 + w * 64 + jj * 16 + l15] = (h16)xv;
      }
  }
  __syncthreads();

  // --- GEMM2: [16,512] x [512,256] -> sC ---
  {
    v8f acc[2] = {};
    for (int k0 = 0; k0 < 512; k0 += 32) {
      v16h af = load_frag(sB + l15 * 512 + k0, hl);
#pragma unroll
      for (int jj = 0; jj < 2; ++jj) {
        const h16* brow = w2t + (size_t)(w * 32 + jj * 16 + l15) * 512 + k0;
        if (k0 + 32 < 512) __builtin_prefetch(brow + 32, 0, 1);
        v16h bf = load_frag(brow, hl);
        acc[jj] = wmma_f16(af, bf, acc[jj]);
      }
    }
#pragma unroll
    for (int jj = 0; jj < 2; ++jj)
#pragma unroll
      for (int r = 0; r < 8; ++r)
        sC[(r + 8 * hl) * 260 + w * 32 + jj * 16 + l15] = acc[jj][r];
  }
  __syncthreads();

  // --- LayerNorm stats: 16 threads per row ---
  {
    int r = t >> 4, sub = t & 15;
    float sm = 0.f, sq = 0.f;
    const float* row = sC + r * 260;
#pragma unroll
    for (int i = 0; i < 16; ++i) {
      float v = row[sub * 16 + i];
      sm += v; sq += v * v;
    }
#pragma unroll
    for (int off = 1; off < 16; off <<= 1) {
      sm += __shfl_xor(sm, off);
      sq += __shfl_xor(sq, off);
    }
    if (sub == 0) {
      float mu = sm * (1.f / 256.f);
      float var = sq * (1.f / 256.f) - mu * mu;
      smu[r] = mu;
      srstd[r] = rsqrtf(var + LN_EPS);
    }
  }
  __syncthreads();

  // --- normalize + residual + NCHW store (pairs of lanes share a channel) ---
#pragma unroll
  for (int pass = 0; pass < 2; ++pass) {
    int c  = (t >> 1) + pass * 128;
    int ph = (t & 1) * 8;
    float g = gamma[c], be = beta[c];
    size_t gb = ((size_t)(b * NC + c) * NH + y) * NW + x0 + ph;
#pragma unroll
    for (int q = 0; q < 8; q += 4) {
      float4 hres = *(const float4*)(xin + gb + q);
      float4 o;
      o.x = hres.x + (sC[(ph + q + 0) * 260 + c] - smu[ph + q + 0]) * srstd[ph + q + 0] * g + be;
      o.y = hres.y + (sC[(ph + q + 1) * 260 + c] - smu[ph + q + 1]) * srstd[ph + q + 1] * g + be;
      o.z = hres.z + (sC[(ph + q + 2) * 260 + c] - smu[ph + q + 2]) * srstd[ph + q + 2] * g + be;
      o.w = hres.w + (sC[(ph + q + 3) * 260 + c] - smu[ph + q + 3]) * srstd[ph + q + 3] * g + be;
      *(float4*)(out + gb + q) = o;
    }
  }
}

// ---------------------------------------------------------------------------
extern "C" void kernel_launch(void* const* d_in, const int* in_sizes, int n_in,
                              void* d_out, int out_size, void* d_ws,
                              size_t ws_size, hipStream_t stream) {
  (void)in_sizes; (void)n_in; (void)out_size; (void)ws_size;
  const float* hidden = (const float*)d_in[0];
  const float* cosm   = (const float*)d_in[1];
  const float* sinm   = (const float*)d_in[2];
  const float* qaggw  = (const float*)d_in[3];
  const float* agg_g  = (const float*)d_in[4];
  const float* agg_b  = (const float*)d_in[5];
  const float* qw     = (const float*)d_in[6];
  const float* kw     = (const float*)d_in[7];
  const float* vw     = (const float*)d_in[8];
  const float* ow     = (const float*)d_in[9];
  const float* fc1w   = (const float*)d_in[10];
  const float* fc2w   = (const float*)d_in[11];
  const float* mlp_g  = (const float*)d_in[12];
  const float* mlp_b  = (const float*)d_in[13];
  float* out = (float*)d_out;

  char* ws = (char*)d_ws;
  size_t off = 0;
  auto take = [&](size_t bytes) {
    char* ptr = ws + off;
    off = (off + bytes + 255) & ~(size_t)255;
    return ptr;
  };

  h16* wq_t = (h16*)take((size_t)256 * 256 * 2);
  h16* wk_t = (h16*)take((size_t)256 * 256 * 2);
  h16* wv_t = (h16*)take((size_t)256 * 256 * 2);
  h16* wo_t = (h16*)take((size_t)256 * 256 * 2);
  h16* w1_t = (h16*)take((size_t)512 * 512 * 2);
  h16* w2_t = (h16*)take((size_t)256 * 512 * 2);
  float* qmap  = (float*)take((size_t)NTOK * NC * 4);
  float* kvmap = (float*)take((size_t)NTOK * NC * 4);
  h16* q_in  = (h16*)take((size_t)NTOK * NC * 2);
  h16* kv_in = (h16*)take((size_t)NTOK * NC * 2);
  float* qs  = (float*)take((size_t)NTOK * NC * 4);
  float* ksb = (float*)take((size_t)NTOK * NC * 4);
  float* vsb = (float*)take((size_t)NTOK * NC * 4);
  h16* Qh = (h16*)take((size_t)NTOK * NC * 2);
  h16* Kh = (h16*)take((size_t)NTOK * NC * 2);
  h16* Vt = (h16*)take((size_t)NTOK * NC * 2);
  h16* ctx = (h16*)take((size_t)NTOK * NC * 2);
  float* msg = (float*)take((size_t)NTOK * NC * 4);

  // weight prep
  k_wt<<<256, 256, 0, stream>>>(qw, wq_t, 256, 256);
  k_wt<<<256, 256, 0, stream>>>(kw, wk_t, 256, 256);
  k_wt<<<256, 256, 0, stream>>>(vw, wv_t, 256, 256);
  k_wt<<<256, 256, 0, stream>>>(ow, wo_t, 256, 256);
  k_wt<<<512, 256, 0, stream>>>(fc1w, w1_t, 512, 512);
  k_wt<<<256, 256, 0, stream>>>(fc2w, w2_t, 512, 256);

  // aggregation + LN
  k_agg<<<dim3(NB * NC, AH), 160, 0, stream>>>(hidden, qaggw, qmap, kvmap);
  k_lnh<<<dim3(NTOK, 2), 256, 0, stream>>>(qmap, kvmap, agg_g, agg_b, q_in, kv_in);

  // QKV projections
  k_gemm<<<dim3(NTOK / 16, NC / 64), 32, 0, stream>>>(q_in,  wq_t, qs,  NTOK, NC, NC);
  k_gemm<<<dim3(NTOK / 16, NC / 64), 32, 0, stream>>>(kv_in, wk_t, ksb, NTOK, NC, NC);
  k_gemm<<<dim3(NTOK / 16, NC / 64), 32, 0, stream>>>(kv_in, wv_t, vsb, NTOK, NC, NC);

  // RoPE + head-major repack
  k_rope<<<NTOK, 128, 0, stream>>>(qs, ksb, vsb, cosm, sinm, Qh, Kh, Vt);

  // flash attention
  k_attn<<<dim3(NS / 64, NB * NHEAD), 128, 0, stream>>>(Qh, Kh, Vt, ctx);

  // output projection -> message
  k_gemm<<<dim3(NTOK / 16, NC / 64), 32, 0, stream>>>(ctx, wo_t, msg, NTOK, NC, NC);

  // fused upsample + MLP + LN + residual
  k_mlp<<<NPIX / 16, 256, 0, stream>>>(hidden, msg, w1_t, w2_t, mlp_g, mlp_b, out);
}
